// _MockMoeLayer_80564996538419
// MI455X (gfx1250) — matrix operand order repaired and checked by reading, not compiled, not run on hardware
//
#include <hip/hip_runtime.h>
#include <math.h>

#define NE   8      // experts
#define TOPK 2
#define HD   2048   // hidden
#define ID   1024   // intermediate
#define NT   4096   // tokens = B*S

typedef __attribute__((ext_vector_type(16))) __bf16 v16bf;
typedef __attribute__((ext_vector_type(8)))  float  v8f;
typedef __attribute__((ext_vector_type(4)))  unsigned v4u;
typedef __attribute__((ext_vector_type(8)))  int v8i;
typedef __attribute__((ext_vector_type(4)))  int v4i;

union FragB16 { v16bf v; uint4 q[2]; };

#if defined(__has_builtin)
#  if __has_builtin(__builtin_amdgcn_tensor_load_to_lds)
#    define HAVE_TDM 1
#  else
#    define HAVE_TDM 0
#  endif
#else
#  define HAVE_TDM 0
#endif

__device__ __forceinline__ unsigned short bf16_bits(float f) {
    union { float f; unsigned u; } c; c.f = f;
    unsigned r = c.u + 0x7FFFu + ((c.u >> 16) & 1u);   // round-to-nearest-even
    return (unsigned short)(r >> 16);
}
__device__ __forceinline__ unsigned pack2(float lo, float hi) {
    return (unsigned)bf16_bits(lo) | ((unsigned)bf16_bits(hi) << 16);
}

#if HAVE_TDM
// ---------------------------------------------------------------------------
// TDM 2D tile load: global (row-major, dim0 contiguous) -> LDS, bf16 elements.
// D# per CDNA5 ISA ch.8. 2D tensors: descriptor groups 2/3 zero.
// ---------------------------------------------------------------------------
__device__ __forceinline__ void tdm_load_2d(
    unsigned lds_off, const void* gptr,
    unsigned tensor_d0, unsigned tensor_d1,
    unsigned tile_d0, unsigned tile_d1,
    unsigned stride_d0_elems)
{
    const unsigned long long ga = (unsigned long long)gptr;
    v4u g0;
    g0.x = 1u;                                                     // count=1, user desc
    g0.y = (unsigned)__builtin_amdgcn_readfirstlane((int)lds_off); // lds_addr
    g0.z = (unsigned)__builtin_amdgcn_readfirstlane((int)(unsigned)ga);
    g0.w = (unsigned)__builtin_amdgcn_readfirstlane(
               (int)(((unsigned)(ga >> 32) & 0x1FFFFFFu) | (2u << 30))); // addr[56:32]|type=2
    v8i g1;
    g1[0] = (int)(1u << 16);                                 // data_size = 2 bytes
    g1[1] = (int)((tensor_d0 & 0xFFFFu) << 16);
    g1[2] = (int)((tensor_d0 >> 16) | ((tensor_d1 & 0xFFFFu) << 16));
    g1[3] = (int)((tensor_d1 >> 16) | (tile_d0 << 16));
    g1[4] = (int)tile_d1;
    g1[5] = (int)stride_d0_elems;
    g1[6] = 0;
    g1[7] = 0;
    v4i z4 = {0, 0, 0, 0};
#if __clang_major__ >= 23
    v8i z8 = {0, 0, 0, 0, 0, 0, 0, 0};
    __builtin_amdgcn_tensor_load_to_lds(g0, g1, z4, z4, z8, 0);
#else
    __builtin_amdgcn_tensor_load_to_lds(g0, g1, z4, z4, 0);
#endif
}
#endif // HAVE_TDM

// ---------------------------------------------------------------------------
// 0) fp32 -> bf16 elementwise conversion (once per launch, memory-bound)
// ---------------------------------------------------------------------------
__global__ __launch_bounds__(256) void cvt_f32_bf16(
    const float* __restrict__ src, unsigned short* __restrict__ dst, int n4)
{
    const int i = blockIdx.x * 256 + threadIdx.x;
    if (i < n4) {
        const float4 v = ((const float4*)src)[i];
        uint2 o; o.x = pack2(v.x, v.y); o.y = pack2(v.z, v.w);
        ((uint2*)dst)[i] = o;
    }
}

// ---------------------------------------------------------------------------
// 1) Router: fp32 logits, softmax, top-2, renorm, atomic append per expert.
// ---------------------------------------------------------------------------
__global__ __launch_bounds__(256) void moe_router(
    const float* __restrict__ x, const float* __restrict__ gate_w,
    int* __restrict__ counts, int* __restrict__ toklist, float* __restrict__ wlist)
{
    const int t   = blockIdx.x;
    const int tid = threadIdx.x;

    float acc[NE];
    #pragma unroll
    for (int e = 0; e < NE; ++e) acc[e] = 0.0f;

    const float* xr = x + (long)t * HD;
    for (int h = tid; h < HD; h += 256) {
        const float xv = xr[h];
        #pragma unroll
        for (int e = 0; e < NE; ++e) acc[e] += xv * gate_w[e * HD + h];
    }

    __shared__ float red[NE][256];
    #pragma unroll
    for (int e = 0; e < NE; ++e) red[e][tid] = acc[e];
    __syncthreads();
    for (int s = 128; s > 0; s >>= 1) {
        if (tid < s) {
            #pragma unroll
            for (int e = 0; e < NE; ++e) red[e][tid] += red[e][tid + s];
        }
        __syncthreads();
    }

    if (tid == 0) {
        float l[NE], p[NE];
        float m = red[0][0];
        #pragma unroll
        for (int e = 0; e < NE; ++e) { l[e] = red[e][0]; m = fmaxf(m, l[e]); }
        float s = 0.0f;
        #pragma unroll
        for (int e = 0; e < NE; ++e) { p[e] = __expf(l[e] - m); s += p[e]; }
        const float inv_s = 1.0f / s;
        #pragma unroll
        for (int e = 0; e < NE; ++e) p[e] *= inv_s;

        int i0 = 0;
        #pragma unroll
        for (int e = 1; e < NE; ++e) if (p[e] > p[i0]) i0 = e;
        int i1 = (i0 == 0) ? 1 : 0;
        #pragma unroll
        for (int e = 0; e < NE; ++e) if (e != i1 && e != i0 && p[e] > p[i1]) i1 = e;

        const float inv = 1.0f / (p[i0] + p[i1]);
        int p0 = atomicAdd(&counts[i0], 1);
        toklist[i0 * NT + p0] = t;  wlist[i0 * NT + p0] = p[i0] * inv;
        int p1 = atomicAdd(&counts[i1], 1);
        toklist[i1 * NT + p1] = t;  wlist[i1 * NT + p1] = p[i1] * inv;
    }
}

// ---------------------------------------------------------------------------
// 2) Prefix sum -> CSR row offsets (sum == NT*TOPK)
// ---------------------------------------------------------------------------
__global__ void moe_prefix(const int* __restrict__ counts, int* __restrict__ offsets)
{
    if (threadIdx.x == 0 && blockIdx.x == 0) {
        int s = 0;
        #pragma unroll
        for (int e = 0; e < NE; ++e) { offsets[e] = s; s += counts[e]; }
        offsets[NE] = s;
    }
}

// ---------------------------------------------------------------------------
// 3) Fused gate+up + SwiGLU. bf16, TDM B tiles, double-buffered.
//    Block = 256 thr (8 waves, 2x4), tile BM=32 x BN=256, BK=64.
//    All 8 B-fragments held in distinct VGPRs so ds_load latency overlaps
//    wmma issue (staged s_wait_dscnt instead of 0 before every wmma).
// ---------------------------------------------------------------------------
__global__ __launch_bounds__(256) void moe_gateup(
    const unsigned short* __restrict__ xb,
    const unsigned short* __restrict__ gate_b,
    const unsigned short* __restrict__ up_b,
    const int* __restrict__ offsets, const int* __restrict__ toklist,
    unsigned short* __restrict__ a_buf)
{
    const int b     = blockIdx.x;
    const int e     = b >> 9;            // 128 mtiles * 4 ntiles
    const int rem   = b & 511;
    const int mtile = rem >> 2;
    const int ntile = rem & 3;
    const int base  = offsets[e];
    const int count = offsets[e + 1] - base;
    const int m0    = mtile * 32;
    if (m0 >= count) return;
    const int n0 = ntile * 256;

    __shared__ __align__(16) unsigned short sA[2][32 * 64];   //  8 KB
    __shared__ __align__(16) unsigned short sG[2][256 * 64];  // 64 KB
    __shared__ __align__(16) unsigned short sU[2][256 * 64];  // 64 KB

    const int tid  = threadIdx.x;
    const int lane = tid & 31;
    const int wave = tid >> 5;
    const int wm   = wave >> 2;
    const int wn   = wave & 3;

    const int  arow   = tid >> 3;        // 0..31
    const int  acol   = (tid & 7) * 8;   // 0..56 (8 bf16 = 16B per thread)
    const int  aslot  = m0 + arow;
    const bool avalid = aslot < count;
    const long atok   = avalid ? (long)toklist[e * NT + aslot] : 0L;
    const unsigned short* xrow = xb + atok * (long)HD + acol;

    const unsigned short* gW = gate_b + ((long)e * ID + n0) * HD;
    const unsigned short* uW = up_b   + ((long)e * ID + n0) * HD;

    v8f cg[4], cu[4];
    #pragma unroll
    for (int i = 0; i < 4; ++i)
        #pragma unroll
        for (int r = 0; r < 8; ++r) { cg[i][r] = 0.0f; cu[i][r] = 0.0f; }

    auto stageA = [&](int i, int buf) {
        uint4 av; av.x = 0u; av.y = 0u; av.z = 0u; av.w = 0u;
        if (avalid) av = *(const uint4*)(xrow + i * 64);
        *(uint4*)&sA[buf][arow * 64 + acol] = av;
    };
    auto stageB = [&](int i, int buf) {
#if HAVE_TDM
        if (tid < 32) {    // wave 0 issues the DMA (EXEC-independent, per-wave cnt)
            tdm_load_2d((unsigned)(unsigned long long)&sG[buf][0], gW + i * 64,
                        HD, ID, 64, 256, HD);
            tdm_load_2d((unsigned)(unsigned long long)&sU[buf][0], uW + i * 64,
                        HD, ID, 64, 256, HD);
        }
#else
        const uint4* gs = (const uint4*)(gW + (long)tid * HD + i * 64);
        const uint4* us = (const uint4*)(uW + (long)tid * HD + i * 64);
        uint4* gd = (uint4*)&sG[buf][tid * 64];
        uint4* ud = (uint4*)&sU[buf][tid * 64];
        #pragma unroll
        for (int j = 0; j < 8; ++j) { gd[j] = gs[j]; ud[j] = us[j]; }
#endif
    };

    stageB(0, 0);
    stageA(0, 0);
    const int NIT = HD / 64;     // 32
    for (int i = 0; i < NIT; ++i) {
        const int cur = i & 1, nxt = cur ^ 1;
        __syncthreads();                       // buf[nxt] reads (iter i-1) drained
        if (i + 1 < NIT) { stageB(i + 1, nxt); stageA(i + 1, nxt); }
#if HAVE_TDM
        if (tid < 32) {
            if (i + 1 < NIT) __builtin_amdgcn_s_wait_tensorcnt(2); // pair(i) done
            else             __builtin_amdgcn_s_wait_tensorcnt(0);
        }
#endif
        __syncthreads();

        const int ar = wm * 16 + (lane & 15);
        const int ac = (lane >= 16) ? 16 : 0;
        const int bc = (lane >= 16) ? 16 : 0;
        #pragma unroll
        for (int kk = 0; kk < 2; ++kk) {
            const int ko = kk * 32;
            FragB16 af;
            af.q[0] = *(const uint4*)&sA[cur][ar * 64 + ko + ac];
            af.q[1] = *(const uint4*)&sA[cur][ar * 64 + ko + ac + 8];
            FragB16 bg[4], bu[4];
            #pragma unroll
            for (int nf = 0; nf < 4; ++nf) {
                const int br = (wn * 64 + nf * 16 + (lane & 15)) * 64 + ko + bc;
                bg[nf].q[0] = *(const uint4*)&sG[cur][br];
                bg[nf].q[1] = *(const uint4*)&sG[cur][br + 8];
                bu[nf].q[0] = *(const uint4*)&sU[cur][br];
                bu[nf].q[1] = *(const uint4*)&sU[cur][br + 8];
            }
            #pragma unroll
            for (int nf = 0; nf < 4; ++nf) {
                cg[nf] = __builtin_amdgcn_wmma_f32_16x16x32_bf16(
                    false, af.v, false, bg[nf].v, (short)0, cg[nf], false, false);
                cu[nf] = __builtin_amdgcn_wmma_f32_16x16x32_bf16(
                    false, af.v, false, bu[nf].v, (short)0, cu[nf], false, false);
            }
        }
    }

    const int mbase = m0 + wm * 16 + ((lane >= 16) ? 8 : 0);
    const int ncol  = n0 + wn * 64 + (lane & 15);
    #pragma unroll
    for (int nf = 0; nf < 4; ++nf) {
        const int ng = ncol + nf * 16;
        #pragma unroll
        for (int r = 0; r < 8; ++r) {
            const int s = mbase + r;
            if (s < count) {
                const float g = cg[nf][r];
                const float u = cu[nf][r];
                const float a = (g / (1.0f + __expf(-g))) * u;
                a_buf[(long)(base + s) * ID + ng] = bf16_bits(a);
            }
        }
    }
}

// ---------------------------------------------------------------------------
// 4) Down projection + routed-weight atomic scatter. Same pipeline shape.
// ---------------------------------------------------------------------------
__global__ __launch_bounds__(256) void moe_down(
    const unsigned short* __restrict__ a_buf,
    const unsigned short* __restrict__ down_b,
    const int* __restrict__ offsets, const int* __restrict__ toklist,
    const float* __restrict__ wlist,
    float* __restrict__ out)
{
    const int b     = blockIdx.x;
    const int e     = b >> 10;           // 128 mtiles * 8 ntiles
    const int rem   = b & 1023;
    const int mtile = rem >> 3;
    const int ntile = rem & 7;
    const int base  = offsets[e];
    const int count = offsets[e + 1] - base;
    const int m0    = mtile * 32;
    if (m0 >= count) return;
    const int n0 = ntile * 256;

    __shared__ __align__(16) unsigned short sA[2][32 * 64];   //  8 KB
    __shared__ __align__(16) unsigned short sB[2][256 * 64];  // 64 KB

    const int tid  = threadIdx.x;
    const int lane = tid & 31;
    const int wave = tid >> 5;
    const int wm   = wave >> 2;
    const int wn   = wave & 3;

    const int  arow   = tid >> 3;
    const int  acol   = (tid & 7) * 8;
    const int  aslot  = m0 + arow;
    const bool avalid = aslot < count;
    const unsigned short* ap =
        a_buf + (long)(base + (avalid ? aslot : 0)) * ID + acol;
    const unsigned short* dW = down_b + ((long)e * HD + n0) * ID;

    v8f c[4];
    #pragma unroll
    for (int i = 0; i < 4; ++i)
        #pragma unroll
        for (int r = 0; r < 8; ++r) c[i][r] = 0.0f;

    auto stageA = [&](int i, int buf) {
        uint4 av; av.x = 0u; av.y = 0u; av.z = 0u; av.w = 0u;
        if (avalid) av = *(const uint4*)(ap + i * 64);
        *(uint4*)&sA[buf][arow * 64 + acol] = av;
    };
    auto stageB = [&](int i, int buf) {
#if HAVE_TDM
        if (tid < 32) {
            tdm_load_2d((unsigned)(unsigned long long)&sB[buf][0], dW + i * 64,
                        ID, HD, 64, 256, ID);
        }
#else
        const uint4* ds_ = (const uint4*)(dW + (long)tid * ID + i * 64);
        uint4* dd = (uint4*)&sB[buf][tid * 64];
        #pragma unroll
        for (int j = 0; j < 8; ++j) dd[j] = ds_[j];
#endif
    };

    stageB(0, 0);
    stageA(0, 0);
    const int NIT = ID / 64;     // 16
    for (int i = 0; i < NIT; ++i) {
        const int cur = i & 1, nxt = cur ^ 1;
        __syncthreads();
        if (i + 1 < NIT) { stageB(i + 1, nxt); stageA(i + 1, nxt); }
#if HAVE_TDM
        if (tid < 32) {
            if (i + 1 < NIT) __builtin_amdgcn_s_wait_tensorcnt(1); // desc(i) done
            else             __builtin_amdgcn_s_wait_tensorcnt(0);
        }
#endif
        __syncthreads();

        const int ar = wm * 16 + (lane & 15);
        const int ac = (lane >= 16) ? 16 : 0;
        const int bc = (lane >= 16) ? 16 : 0;
        #pragma unroll
        for (int kk = 0; kk < 2; ++kk) {
            const int ko = kk * 32;
            FragB16 af;
            af.q[0] = *(const uint4*)&sA[cur][ar * 64 + ko + ac];
            af.q[1] = *(const uint4*)&sA[cur][ar * 64 + ko + ac + 8];
            FragB16 bb[4];
            #pragma unroll
            for (int nf = 0; nf < 4; ++nf) {
                const int br = (wn * 64 + nf * 16 + (lane & 15)) * 64 + ko + bc;
                bb[nf].q[0] = *(const uint4*)&sB[cur][br];
                bb[nf].q[1] = *(const uint4*)&sB[cur][br + 8];
            }
            #pragma unroll
            for (int nf = 0; nf < 4; ++nf) {
                c[nf] = __builtin_amdgcn_wmma_f32_16x16x32_bf16(
                    false, af.v, false, bb[nf].v, (short)0, c[nf], false, false);
            }
        }
    }

    const int mbase = m0 + wm * 16 + ((lane >= 16) ? 8 : 0);
    const int ncol  = n0 + wn * 64 + (lane & 15);
    #pragma unroll
    for (int nf = 0; nf < 4; ++nf) {
        const int ng = ncol + nf * 16;
        #pragma unroll
        for (int r = 0; r < 8; ++r) {
            const int s = mbase + r;
            if (s < count) {
                const int   tok = toklist[e * NT + s];
                const float w   = wlist[e * NT + s];
                atomicAdd(&out[(long)tok * HD + ng], w * c[nf][r]);
            }
        }
    }
}

// ---------------------------------------------------------------------------
// Host launcher (graph-capture safe). Workspace (~128.3 MB):
//   [0,64) counts   [64,128) offsets   [128..) toklist / wlist (NE*NT each)
//   @262400: xb bf16 [NT,HD] (16MB); gate_b/up_b bf16 [NE,ID,HD] (32MB each);
//   down_b bf16 [NE,HD,ID] (32MB); a_buf bf16 [NT*TOPK, ID] (16MB)
// ---------------------------------------------------------------------------
extern "C" void kernel_launch(void* const* d_in, const int* in_sizes, int n_in,
                              void* d_out, int out_size, void* d_ws, size_t ws_size,
                              hipStream_t stream)
{
    const float* x         = (const float*)d_in[0];
    const float* gate_w    = (const float*)d_in[1];
    const float* gate_proj = (const float*)d_in[2];
    const float* up_proj   = (const float*)d_in[3];
    const float* down_proj = (const float*)d_in[4];
    float* out = (float*)d_out;

    char* ws = (char*)d_ws;
    int*   counts  = (int*)(ws);
    int*   offsets = (int*)(ws + 64);
    int*   toklist = (int*)(ws + 128);
    float* wlist   = (float*)(ws + 128 + (size_t)NE * NT * 4);
    unsigned short* xb = (unsigned short*)(ws + 262400);
    unsigned short* gb = xb + (size_t)NT * HD;
    unsigned short* ub = gb + (size_t)NE * ID * HD;
    unsigned short* db = ub + (size_t)NE * ID * HD;
    unsigned short* ab = db + (size_t)NE * HD * ID;

    hipMemsetAsync(counts, 0, 64, stream);
    hipMemsetAsync(out, 0, (size_t)NT * HD * sizeof(float), stream);

    const int nx4 = NT * HD / 4;          // 2,097,152
    const int nw4 = NE * ID * HD / 4;     // 4,194,304
    cvt_f32_bf16<<<(nx4 + 255) / 256, 256, 0, stream>>>(x, xb, nx4);
    cvt_f32_bf16<<<(nw4 + 255) / 256, 256, 0, stream>>>(gate_proj, gb, nw4);
    cvt_f32_bf16<<<(nw4 + 255) / 256, 256, 0, stream>>>(up_proj, ub, nw4);
    cvt_f32_bf16<<<(nw4 + 255) / 256, 256, 0, stream>>>(down_proj, db, nw4);

    moe_router<<<NT, 256, 0, stream>>>(x, gate_w, counts, toklist, wlist);
    moe_prefix<<<1, 32, 0, stream>>>(counts, offsets);
    moe_gateup<<<NE * 128 * 4, 256, 0, stream>>>(xb, gb, ub, offsets, toklist, ab);
    moe_down<<<NE * 128 * 8, 256, 0, stream>>>(ab, db, offsets, toklist, wlist, out);
}
